// LSTM_91216515432820
// MI455X (gfx1250) — matrix operand order repaired and checked by reading, not compiled, not run on hardware
//
#include <hip/hip_runtime.h>
#include <hip/hip_bf16.h>

typedef unsigned short u16;
typedef __attribute__((ext_vector_type(16))) __bf16 v16bf;
typedef __attribute__((ext_vector_type(8)))  float  v8f;
typedef __attribute__((ext_vector_type(4)))  unsigned int v4u;
typedef __attribute__((ext_vector_type(8)))  int v8i;
typedef __attribute__((ext_vector_type(4)))  int v4i;

#define NWG   64        // persistent workgroups (one per WGP), each owns 16 latent dims
#define TLEN  512
#define BATCH 64
#define DL    1024
#define DI    512
#define KFULL 1536      // DL + DI (packed weight row count)
#define NCOL  4096      // 4 gates * DL

// LDS row strides padded +8 u16 -> rows land 4 banks apart, conflict-free ds_load_b128
#define MS_XW    (DL + 8)                     // 1032: recurrent stages h only
#define MS_FU    (KFULL + 8)                  // 1544: fused stages [h|x]
#define MS_XE    (DI + 8)                     // 520:  x-precompute stages x rows
#define XWL_U16  (BATCH * 64)                 // one XW gate block: 8 KB
#define SMEM_LSTM_XW (BATCH*MS_XW*2 + BATCH*64*4 + BATCH*16*4 + 2*XWL_U16*2) // 168960
#define SMEM_LSTM_FU (BATCH*MS_FU*2 + BATCH*64*4 + BATCH*16*4)               // 218112
#define SMEM_XWK     (BATCH*MS_XE*2)                                         // 66560

union Frag { uint4 u[2]; v16bf v; };

__device__ __forceinline__ u16 f2bf(float f) {
  unsigned int u = __float_as_uint(f);
  return (u16)((u + 0x7FFFu + ((u >> 16) & 1u)) >> 16);     // RNE
}
__device__ __forceinline__ float b2f(u16 v) {
  return __uint_as_float((unsigned int)v << 16);
}
__device__ __forceinline__ float sigf(float x)  { return 1.0f / (1.0f + __expf(-x)); }
__device__ __forceinline__ float tanhf_(float x){ return 2.0f * sigf(2.0f * x) - 1.0f; }

// ---------------- one-time prep kernels ----------------

__global__ void k_init(const float* __restrict__ h, u16* __restrict__ hbuf,
                       int* __restrict__ bar) {
  int i = blockIdx.x * 256 + threadIdx.x;
  if (i < BATCH * DL) hbuf[i] = f2bf(h[i]);
  if (i == 0) { bar[0] = 0; bar[1] = 0; }
}

// Wp[n*1536 + k] = bf16(W_gate[k][J]); col n: wg=n>>6, gate=(n>>4)&3, J=wg*16+(n&15)
// k rows 0..1023 = h part, 1024..1535 = x part (matches m = [h|x])
__global__ void k_pack(const float* __restrict__ Wf, const float* __restrict__ Wi,
                       const float* __restrict__ Wc, const float* __restrict__ Wo,
                       const float* __restrict__ bf_, const float* __restrict__ bi_,
                       const float* __restrict__ bc_, const float* __restrict__ bo_,
                       u16* __restrict__ Wp, float* __restrict__ bp) {
  int i = blockIdx.x * 256 + threadIdx.x;          // chunk: (n, k0/8)
  if (i >= NCOL * (KFULL / 8)) return;
  int n = i / (KFULL / 8), k0 = (i % (KFULL / 8)) * 8;
  int wgi = n >> 6, gate = (n >> 4) & 3, jl = n & 15;
  int J = wgi * 16 + jl;
  const float* W = gate == 0 ? Wf : gate == 1 ? Wi : gate == 2 ? Wc : Wo;
  const float* b = gate == 0 ? bf_ : gate == 1 ? bi_ : gate == 2 ? bc_ : bo_;
  alignas(16) u16 o[8];
#pragma unroll
  for (int r = 0; r < 8; ++r) o[r] = f2bf(W[(size_t)(k0 + r) * DL + J]);
  *(uint4*)(Wp + (size_t)n * KFULL + k0) = *(const uint4*)o;
  if (k0 == 0) bp[n] = b[J];
}

__global__ void k_embed(const int* __restrict__ X, const float* __restrict__ emb,
                        u16* __restrict__ Xe) {
  int i = blockIdx.x * 256 + threadIdx.x;          // chunk: (t*b, d0/8)
  if (i >= TLEN * BATCH * (DI / 8)) return;
  int tb = i >> 6, c8 = (i & 63) * 8;
  int tok = X[tb];
  const float* s = emb + (size_t)tok * DI + c8;
  alignas(16) u16 o[8];
#pragma unroll
  for (int r = 0; r < 8; ++r) o[r] = f2bf(s[r]);
  *(uint4*)(Xe + (size_t)tb * DI + c8) = *(const uint4*)o;
}

// ---------------- parallel x-projection: XW[t][wg][row][4x16] = x_t @ Wx ----------------

__global__ void __launch_bounds__(256, 1)
k_xw(const u16* __restrict__ Xe, const u16* __restrict__ Wp, u16* __restrict__ XW) {
  extern __shared__ char smem[];
  u16* xl = (u16*)smem;                            // [64][MS_XE]
  const int tid = threadIdx.x;
  const int t = blockIdx.x >> 6, nb = blockIdx.x & 63;
  const int wave = tid >> 5, lane = tid & 31, half = lane >> 4, l15 = lane & 15;
  const int strip = wave >> 1, mh = wave & 1;

  const u16* xet = Xe + (size_t)t * BATCH * DI;
  for (int i = tid; i < BATCH * (DI / 8); i += 256) {
    int row = i >> 6, cu = (i & 63) * 8;
    *(uint4*)(xl + row * MS_XE + cu) = *(const uint4*)(xet + (size_t)row * DI + cu);
  }
  __syncthreads();

  const u16* wcol = Wp + (size_t)(nb * 64 + strip * 16 + l15) * KFULL + DL; // x rows
  v8f acc0 = {}, acc1 = {};
#pragma unroll 2
  for (int kt = 0; kt < DI / 32; ++kt) {
    const int kb = kt * 32;
    Frag b;
    b.u[0] = *(const uint4*)(wcol + kb + half * 16);
    b.u[1] = *(const uint4*)(wcol + kb + half * 16 + 8);
    const u16* ab = xl + kb + half * 16;
    const int r0 = mh * 32;
    Frag a0, a1;
    a0.u[0] = *(const uint4*)(ab + (r0 + l15) * MS_XE);
    a0.u[1] = *(const uint4*)(ab + (r0 + l15) * MS_XE + 8);
    a1.u[0] = *(const uint4*)(ab + (r0 + 16 + l15) * MS_XE);
    a1.u[1] = *(const uint4*)(ab + (r0 + 16 + l15) * MS_XE + 8);
    acc0 = __builtin_amdgcn_wmma_f32_16x16x32_bf16(false, a0.v, false, b.v, (short)0, acc0, false, false);
    acc1 = __builtin_amdgcn_wmma_f32_16x16x32_bf16(false, a1.v, false, b.v, (short)0, acc1, false, false);
  }

  u16* out = XW + (((size_t)t * NWG + nb) * BATCH) * 64;
  const int col = strip * 16 + l15;
  const int r0 = mh * 32 + half * 8;
#pragma unroll
  for (int r = 0; r < 8; ++r) {
    out[(r0 + r) * 64 + col]      = f2bf(acc0[r]);
    out[(r0 + 16 + r) * 64 + col] = f2bf(acc1[r]);
  }
}

// ---------------- persistent recurrent kernel ----------------

__device__ __forceinline__ void grid_sync(int* bar) {
  __threadfence();
  __syncthreads();
  if (threadIdx.x == 0) {
    int* cnt = bar;
    int* gen = bar + 1;
    int g = __hip_atomic_load(gen, __ATOMIC_ACQUIRE, __HIP_MEMORY_SCOPE_AGENT);
    int prev = __hip_atomic_fetch_add(cnt, 1, __ATOMIC_ACQ_REL, __HIP_MEMORY_SCOPE_AGENT);
    if (prev == NWG - 1) {
      __hip_atomic_store(cnt, 0, __ATOMIC_RELAXED, __HIP_MEMORY_SCOPE_AGENT);
      __hip_atomic_fetch_add(gen, 1, __ATOMIC_ACQ_REL, __HIP_MEMORY_SCOPE_AGENT);
    } else {
      while (__hip_atomic_load(gen, __ATOMIC_ACQUIRE, __HIP_MEMORY_SCOPE_AGENT) == g)
        __builtin_amdgcn_s_sleep(1);
    }
  }
  __syncthreads();
}

// TDM: DMA one 8KB XW gate block (2048 dwords, linear) into LDS. ISA 08 §8.3/8.4.
// clang-23 toolchain: 6-arg builtin (g0, g1, g2, g3, g1b, cpol); groups 2/3 zero => <=2D.
__device__ __forceinline__ void tdm_load_xw(unsigned lds_off, const u16* gsrc) {
  unsigned long long ga = (unsigned long long)(size_t)gsrc;
  v4u g0;
  g0[0] = 1u;                                  // count=1, user descriptor
  g0[1] = lds_off;                             // lds_addr (bytes)
  g0[2] = (unsigned)ga;                        // global_addr[31:0]
  g0[3] = (unsigned)(ga >> 32) | (2u << 30);   // global_addr[56:32] | type=2 (image)
  v8i g1;
  g1[0] = 0x20000;                 // workgroup_mask=0, data_size=2 (4-byte units)
  g1[1] = (int)(2048u << 16);      // tensor_dim0 = 2048 dwords (bits 63:48)
  g1[2] = 0;                       // tensor_dim0 hi, tensor_dim1 unused
  g1[3] = (int)(2048u << 16);      // tile_dim0 = 2048 (bits 127:112)
  g1[4] = 0;                       // tile_dim1/2 unused
  g1[5] = 2048;                    // tensor_dim0_stride = 2048 dwords
  g1[6] = 0;
  g1[7] = 0;
  __builtin_amdgcn_tensor_load_to_lds(g0, g1, (v4i)0, (v4i)0, (v8i)0, 0);
}

template <bool USE_XW>
__global__ void __launch_bounds__(256, 1)
k_lstm(const u16* __restrict__ Xe, const u16* __restrict__ XW,
       const u16* __restrict__ Wp, const float* __restrict__ bp,
       u16* __restrict__ hbuf, float* __restrict__ H, int* __restrict__ bar) {
  constexpr int KT  = USE_XW ? DL : KFULL;      // sequential reduction depth
  constexpr int NKT = KT / 32;
  constexpr int MS  = USE_XW ? MS_XW : MS_FU;

  extern __shared__ char smem[];
  u16*   mlds = (u16*)smem;                                       // [64][MS] bf16
  float* gbuf = (float*)(smem + BATCH * MS * 2);                  // [64][64] gate pre-acts
  float* cst  = (float*)(smem + BATCH * MS * 2 + BATCH * 64 * 4); // [64][16] cell state
  u16*   xwl  = (u16*)(smem + BATCH * MS * 2 + BATCH * 64 * 4 + BATCH * 16 * 4); // 2x[64][64]

  const int tid  = threadIdx.x;
  const int wg   = blockIdx.x;
  const int wave = tid >> 5, lane = tid & 31, half = lane >> 4, l15 = lane & 15;
  const int strip = wave >> 1;    // N strip (0..3) of 16 cols
  const int mh    = wave & 1;     // M half (rows 0..31 / 32..63)

  for (int i = tid; i < BATCH * 16; i += 256) cst[i] = 0.0f;

  const u16* wcol = Wp + (size_t)(wg * 64 + strip * 16 + l15) * KFULL;

  const int j = tid & 15;
  const float bF = bp[wg * 64 + j];
  const float bI = bp[wg * 64 + 16 + j];
  const float bC = bp[wg * 64 + 32 + j];
  const float bO = bp[wg * 64 + 48 + j];

  if (USE_XW && wave == 0) {   // prime the TDM pipeline: XW block for t=0 -> buf0
    tdm_load_xw((unsigned)(size_t)(void*)xwl,
                XW + (((size_t)0 * NWG + wg) * BATCH) * 64);
  }
  __syncthreads();

#pragma unroll 1
  for (int t = 0; t < TLEN; ++t) {
    const u16* hprev = hbuf + (size_t)(t & 1) * (BATCH * DL);

    // ---- stage A operand into LDS ----
    if (USE_XW) {                      // h only: 64 x 1024 bf16
      for (int i = tid; i < BATCH * (DL / 8); i += 256) {
        int row = i >> 7, cu = (i & 127) * 8;
        *(uint4*)(mlds + row * MS + cu) = *(const uint4*)(hprev + (size_t)row * DL + cu);
      }
    } else {                           // m = [h | x]: 64 x 1536 bf16
      const u16* xet = Xe + (size_t)t * BATCH * DI;
      for (int i = tid; i < BATCH * (KFULL / 8); i += 256) {
        int row = i / (KFULL / 8), cu = (i % (KFULL / 8)) * 8;
        uint4 v;
        if (cu < DL) v = *(const uint4*)(hprev + (size_t)row * DL + cu);
        else         v = *(const uint4*)(xet + (size_t)row * DI + (cu - DL));
        *(uint4*)(mlds + row * MS + cu) = v;
      }
    }
    __syncthreads();

    // ---- async: DMA next step's XW gate block while the GEMM runs ----
    if (USE_XW && wave == 0 && t + 1 < TLEN) {
      tdm_load_xw((unsigned)(size_t)(void*)xwl + ((t + 1) & 1) * (XWL_U16 * 2),
                  XW + (((size_t)(t + 1) * NWG + wg) * BATCH) * 64);
    }

    // ---- 32x16 strip GEMM over K ----
    v8f acc0 = {}, acc1 = {};
#pragma unroll 2
    for (int kt = 0; kt < NKT; ++kt) {
      const int kb = kt * 32;
      Frag b;
      b.u[0] = *(const uint4*)(wcol + kb + half * 16);
      b.u[1] = *(const uint4*)(wcol + kb + half * 16 + 8);
      __builtin_prefetch(wcol + kb + 256, 0, 3);   // 8 k-steps ahead, into WGP$

      const u16* ab = mlds + kb + half * 16;
      const int r0 = mh * 32;
      Frag a0, a1;
      a0.u[0] = *(const uint4*)(ab + (r0 + l15) * MS);
      a0.u[1] = *(const uint4*)(ab + (r0 + l15) * MS + 8);
      a1.u[0] = *(const uint4*)(ab + (r0 + 16 + l15) * MS);
      a1.u[1] = *(const uint4*)(ab + (r0 + 16 + l15) * MS + 8);

      acc0 = __builtin_amdgcn_wmma_f32_16x16x32_bf16(false, a0.v, false, b.v, (short)0, acc0, false, false);
      acc1 = __builtin_amdgcn_wmma_f32_16x16x32_bf16(false, a1.v, false, b.v, (short)0, acc1, false, false);
    }

    // ---- spill gate pre-acts: C row = mh*32 + tile*16 + half*8 + r, col = strip*16+l15 ----
    {
      const int col = strip * 16 + l15;
      const int r0  = mh * 32 + half * 8;
#pragma unroll
      for (int r = 0; r < 8; ++r) {
        gbuf[(r0 + r) * 64 + col]      = acc0[r];
        gbuf[(r0 + 16 + r) * 64 + col] = acc1[r];
      }
    }
    // wave0 retires the oldest outstanding TDM load (this step's XW block),
    // then the barrier publishes both gbuf and xwl to the whole WG.
    if (USE_XW && wave == 0) {
      if (t + 1 < TLEN) __builtin_amdgcn_s_wait_tensorcnt(1);
      else              __builtin_amdgcn_s_wait_tensorcnt(0);
    }
    __syncthreads();

    // ---- gating ----
    u16*   hnext = hbuf + (size_t)((t + 1) & 1) * (BATCH * DL);
    float* Ht    = H + (size_t)t * (BATCH * DL);
    const u16* xwt = xwl + (t & 1) * XWL_U16;
    for (int i = tid; i < BATCH * 16; i += 256) {
      int row = i >> 4;
      float f  = gbuf[row * 64 + j]      + bF;
      float ig = gbuf[row * 64 + 16 + j] + bI;
      float cc = gbuf[row * 64 + 32 + j] + bC;
      float og = gbuf[row * 64 + 48 + j] + bO;
      if (USE_XW) {
        const u16* xr = xwt + row * 64;
        f += b2f(xr[j]); ig += b2f(xr[16 + j]); cc += b2f(xr[32 + j]); og += b2f(xr[48 + j]);
      }
      float ct = sigf(f) * cst[i] + sigf(ig) * tanhf_(cc);
      float ht = sigf(og) * tanhf_(ct);
      cst[i] = ct;
      int J = wg * 16 + j;
      hnext[(size_t)row * DL + J] = f2bf(ht);
      Ht[(size_t)row * DL + J]    = ht;
    }
    __syncthreads();

    grid_sync(bar);
  }
}

// ---------------- host launcher ----------------

extern "C" void kernel_launch(void* const* d_in, const int* in_sizes, int n_in,
                              void* d_out, int out_size, void* d_ws, size_t ws_size,
                              hipStream_t stream) {
  const int*   X   = (const int*)d_in[0];
  const float* h   = (const float*)d_in[1];
  const float* emb = (const float*)d_in[2];
  const float* Wf  = (const float*)d_in[3];
  const float* bf_ = (const float*)d_in[4];
  const float* Wi  = (const float*)d_in[5];
  const float* bi_ = (const float*)d_in[6];
  const float* Wc  = (const float*)d_in[7];
  const float* bc_ = (const float*)d_in[8];
  const float* Wo  = (const float*)d_in[9];
  const float* bo_ = (const float*)d_in[10];
  float* H = (float*)d_out;

  char* ws = (char*)d_ws;
  size_t off = 0;
  int* bar  = (int*)(ws + off);   off += 256;
  u16* hbuf = (u16*)(ws + off);   off += (size_t)2 * BATCH * DL * 2;     // 256 KB
  u16* Xe   = (u16*)(ws + off);   off += (size_t)TLEN * BATCH * DI * 2;  // 32 MB
  u16* Wp   = (u16*)(ws + off);   off += (size_t)NCOL * KFULL * 2;       // 12.6 MB
  float* bp = (float*)(ws + off); off += (size_t)NCOL * 4;
  off = (off + 255) & ~(size_t)255;
  u16* XW   = (u16*)(ws + off);
  const size_t xw_bytes = (size_t)TLEN * BATCH * NCOL * 2;               // 256 MB
  const bool use_xw = ws_size >= off + xw_bytes;

  (void)hipFuncSetAttribute(reinterpret_cast<const void*>(&k_lstm<true>),
                            hipFuncAttributeMaxDynamicSharedMemorySize, SMEM_LSTM_XW);
  (void)hipFuncSetAttribute(reinterpret_cast<const void*>(&k_lstm<false>),
                            hipFuncAttributeMaxDynamicSharedMemorySize, SMEM_LSTM_FU);
  (void)hipFuncSetAttribute(reinterpret_cast<const void*>(&k_xw),
                            hipFuncAttributeMaxDynamicSharedMemorySize, SMEM_XWK);

  k_init<<<(BATCH * DL + 255) / 256, 256, 0, stream>>>(h, hbuf, bar);
  k_pack<<<(NCOL * (KFULL / 8) + 255) / 256, 256, 0, stream>>>(
      Wf, Wi, Wc, Wo, bf_, bi_, bc_, bo_, Wp, bp);
  k_embed<<<(TLEN * BATCH * (DI / 8) + 255) / 256, 256, 0, stream>>>(X, emb, Xe);

  if (use_xw) {
    k_xw<<<TLEN * NWG, 256, SMEM_XWK, stream>>>(Xe, Wp, XW);
    k_lstm<true><<<NWG, 256, SMEM_LSTM_XW, stream>>>(Xe, XW, Wp, bp, hbuf, H, bar);
  } else {
    k_lstm<false><<<NWG, 256, SMEM_LSTM_FU, stream>>>(Xe, XW, Wp, bp, hbuf, H, bar);
  }
}
